// PointNet_29506425323597
// MI455X (gfx1250) — compile-verified, matrix operand order checked by reading, not run on hardware
//
#include <hip/hip_runtime.h>
#include <hip/hip_bf16.h>

// ---------------------------------------------------------------------------
// PointNet edge-conv for MI455X (gfx1250, wave32, WMMA).
//
// Decomposition: message relu([h_src, dpos] @ Wloc + b) is split into
//   Y = H @ Wloc[:nin]           (dense bf16 WMMA GEMM, per node)
//   m_e = relu(Y[src] + dpos @ Wloc[nin:nin+2] + b)  (per edge, rank-2 fixup)
// max-agg is a clean per-node loop since dst = repeat(arange(N), K) (+ self).
//
// GEMM: each wave32 owns ONE 16x32 A fragment and TWO packed B fragments
// (two 16x16 f32 accumulators) -> 2 WMMA per A load group, all N widths
// padded to 32 so EXEC stays all-ones around the matrix ops.
// ---------------------------------------------------------------------------

#define NNODES 80000
#define KNN    8
#define NBATCH 64

typedef __bf16 bf16_t;
typedef bf16_t v8bf  __attribute__((ext_vector_type(8)));
typedef bf16_t v16bf __attribute__((ext_vector_type(16)));
typedef float  v8f   __attribute__((ext_vector_type(8)));

// K index inside a 32-wide k-tile for the 16-bit A/B WMMA fragment layout
// (ISA 7.12.2): element j of the 16-half fragment, for a given lane.
__device__ __forceinline__ int frag_kmap(int lane, int j) {
    return ((j >> 3) << 4) | (((lane >> 4) & 1) << 3) | (j & 7);
}

// ---------------------------------------------------------------------------
// Pack f32 weight [rows x cols] into WMMA-B fragment order, bf16, zero padded.
// Record layout: ((kt*ntiles16 + nt)*32 + lane)*16 + j   (halves).
// ---------------------------------------------------------------------------
__global__ void pack_w_kernel(bf16_t* __restrict__ dst, const float* __restrict__ W,
                              int rows, int cols, int ktiles, int ntiles16) {
    int idx = blockIdx.x * blockDim.x + threadIdx.x;
    int total = ktiles * ntiles16 * 32 * 16;
    if (idx >= total) return;
    int j    = idx & 15;
    int lane = (idx >> 4) & 31;
    int t    = idx >> 9;
    int nt   = t % ntiles16;
    int kt   = t / ntiles16;
    int k = kt * 32 + frag_kmap(lane, j);
    int n = nt * 16 + (lane & 15);
    float v = (k < rows && n < cols) ? W[(size_t)k * cols + n] : 0.0f;
    dst[idx] = (bf16_t)v;
}

// ---------------------------------------------------------------------------
// f32 -> bf16 elementwise (layer-0 activation pack, stride == 256 == F0).
// ---------------------------------------------------------------------------
__global__ void pack_x_kernel(bf16_t* __restrict__ dst, const float* __restrict__ x,
                              size_t total) {
    size_t i = (size_t)blockIdx.x * blockDim.x + threadIdx.x;
    if (i < total) dst[i] = (bf16_t)x[i];
}

// ---------------------------------------------------------------------------
// f32 [N x ld] (valid cols < ncols) -> bf16 [N x ld], zero padded.
// ---------------------------------------------------------------------------
__global__ __launch_bounds__(256)
void pack_act_kernel(bf16_t* __restrict__ dst, const float* __restrict__ src,
                     int ncols, int ld) {
    int i = blockIdx.x;
    int j = threadIdx.x;
    if (j >= ld) return;
    float v = (j < ncols) ? src[(size_t)i * ld + j] : 0.0f;
    dst[(size_t)i * ld + j] = (bf16_t)v;
}

// ---------------------------------------------------------------------------
// bf16 WMMA GEMM: C[m,n] = sum_k A[m,k] * Bpk[k,n]  (+bias, +relu), f32 out.
// Block = 256 threads = 8 wave32 waves stacked along M (BM = 128); each wave
// computes a 16x32 C strip as two 16x16 WMMA tiles sharing one A fragment.
// grid = (nrows/128, npad/32).   lda = Kpad (mult of 32), ldc = Npad (mult 32)
// ---------------------------------------------------------------------------
__global__ __launch_bounds__(256)
void gemm_wmma_bf16(const bf16_t* __restrict__ A, const bf16_t* __restrict__ Bpk,
                    const float* __restrict__ bias, float* __restrict__ C,
                    int nrows, int ktiles, int ntiles16, int lda, int ldc,
                    int ncols_real, int do_relu) {
    const int lane = threadIdx.x & 31;
    const int wave = threadIdx.x >> 5;
    const int m0   = blockIdx.x * 128 + wave * 16;
    const int nt0  = blockIdx.y * 2;             // two consecutive 16-wide B tiles
    const int nc   = lane & 15;
    const int row  = m0 + nc;
    const int khalf = (lane >> 4) << 3;          // 0 or 8: lane-group K offset

    v8f acc0 = {};
    v8f acc1 = {};
    const bf16_t* aptr = A + (size_t)row * lda + khalf;
    const bf16_t* bptr = Bpk + ((size_t)nt0 * 32 + lane) * 16;
    const size_t  bstep = (size_t)ntiles16 * 32 * 16;

    for (int kt = 0; kt < ktiles; ++kt) {
        v8bf alo = *reinterpret_cast<const v8bf*>(aptr);        // K = {0..7}+khalf
        v8bf ahi = *reinterpret_cast<const v8bf*>(aptr + 16);   // K = {16..23}+khalf
        v8bf b0l = *reinterpret_cast<const v8bf*>(bptr);
        v8bf b0h = *reinterpret_cast<const v8bf*>(bptr + 8);
        v8bf b1l = *reinterpret_cast<const v8bf*>(bptr + 512);  // next 16-wide tile
        v8bf b1h = *reinterpret_cast<const v8bf*>(bptr + 520);
        v16bf a  = __builtin_shufflevector(alo, ahi, 0,1,2,3,4,5,6,7,8,9,10,11,12,13,14,15);
        v16bf b0 = __builtin_shufflevector(b0l, b0h, 0,1,2,3,4,5,6,7,8,9,10,11,12,13,14,15);
        v16bf b1 = __builtin_shufflevector(b1l, b1h, 0,1,2,3,4,5,6,7,8,9,10,11,12,13,14,15);
        acc0 = __builtin_amdgcn_wmma_f32_16x16x32_bf16(
                   false, a, false, b0, (short)0, acc0, false, false);
        acc1 = __builtin_amdgcn_wmma_f32_16x16x32_bf16(
                   false, a, false, b1, (short)0, acc1, false, false);
        aptr += 32;
        bptr += bstep;
    }

    const int n0 = nt0 * 16 + nc;
    const int n1 = n0 + 16;
    float bv0 = (bias != nullptr && n0 < ncols_real) ? bias[n0] : 0.0f;
    float bv1 = (bias != nullptr && n1 < ncols_real) ? bias[n1] : 0.0f;
    const int mb = m0 + ((lane >> 4) << 3);   // C layout: vgpr v -> M = v (+8 hi lanes)
#pragma unroll
    for (int v = 0; v < 8; ++v) {
        int m = mb + v;
        if (m < nrows) {
            float r0 = acc0[v] + bv0;
            float r1 = acc1[v] + bv1;
            if (do_relu) { r0 = fmaxf(r0, 0.0f); r1 = fmaxf(r1, 0.0f); }
            C[(size_t)m * ldc + n0] = r0;
            C[(size_t)m * ldc + n1] = r1;
        }
    }
}

// ---------------------------------------------------------------------------
// Per-node max aggregation over K neighbors + self loop.
// agg[i,j] = max over s in {src[i*K..], i} of relu(Y[s,j] + rel.x*w2x[j]
//                                                 + rel.y*w2y[j] + b[j])
// Block per node, thread per column.  Y gathers are L2-resident (<=72MB).
// ---------------------------------------------------------------------------
__global__ __launch_bounds__(256)
void edge_aggregate_kernel(float* __restrict__ agg, const float* __restrict__ y,
                           const float* __restrict__ pos, const int* __restrict__ src,
                           const float* __restrict__ w2, const float* __restrict__ bias,
                           int ncols, int ldy) {
    const int i = blockIdx.x;
    const int j = threadIdx.x;
    if (j >= ncols) return;
    const float w2x = w2[j];
    const float w2y = w2[ncols + j];
    const float b   = bias[j];
    const float px = pos[2 * i], py = pos[2 * i + 1];
    // self loop: rel = 0
    float best = fmaxf(y[(size_t)i * ldy + j] + b, 0.0f);
#pragma unroll
    for (int k = 0; k < KNN; ++k) {
        int s = src[i * KNN + k];                       // uniform per block -> scalar
        if (k + 1 < KNN) {                              // prefetch next gathered row
            int s2 = src[i * KNN + k + 1];
            __builtin_prefetch(&y[(size_t)s2 * ldy + j], 0, 1);
        }
        float rx = pos[2 * s] - px;
        float ry = pos[2 * s + 1] - py;
        float v = y[(size_t)s * ldy + j] + rx * w2x + ry * w2y + b;
        best = fmaxf(best, fmaxf(v, 0.0f));
    }
    agg[(size_t)i * ldy + j] = best;
}

// ---------------------------------------------------------------------------
// BatchNorm column statistics: two stage (register partials -> f32 atomics).
// ---------------------------------------------------------------------------
__global__ __launch_bounds__(256)
void colstats_kernel(const float* __restrict__ g, float* __restrict__ sums,
                     float* __restrict__ sumsq, int ncols, int ldg,
                     int rows_per_block, int nrows) {
    const int c = threadIdx.x;
    if (c >= ncols) return;
    int r0 = blockIdx.x * rows_per_block;
    int r1 = r0 + rows_per_block;
    if (r1 > nrows) r1 = nrows;
    float s = 0.0f, q = 0.0f;
    for (int r = r0; r < r1; ++r) {
        float v = g[(size_t)r * ldg + c];
        s += v;
        q += v * v;
    }
    atomicAdd(&sums[c], s);
    atomicAdd(&sumsq[c], q);
}

__global__ void bn_finalize_kernel(const float* __restrict__ sums,
                                   const float* __restrict__ sumsq,
                                   float* __restrict__ scale, float* __restrict__ shift,
                                   const float* __restrict__ gamma,
                                   const float* __restrict__ beta,
                                   int ncols, float invn) {
    int c = blockIdx.x * blockDim.x + threadIdx.x;
    if (c >= ncols) return;
    float mu  = sums[c] * invn;
    float var = sumsq[c] * invn - mu * mu;       // population var (jnp.var)
    float rs  = rsqrtf(var + 1e-5f);
    float sc  = gamma[c] * rs;
    scale[c] = sc;
    shift[c] = beta[c] - mu * sc;
}

// ---------------------------------------------------------------------------
// h = relu(g*scale + shift); write bf16 [N x ld] (zero padded, next GEMM A)
// and optionally f32 [N x ld] (for the final pooling pass).
// ---------------------------------------------------------------------------
__global__ __launch_bounds__(256)
void bn_relu_pack_kernel(bf16_t* __restrict__ hout, float* __restrict__ fout,
                         const float* __restrict__ g, const float* __restrict__ scale,
                         const float* __restrict__ shift, int ncols, int ld,
                         int write_f32) {
    const int i = blockIdx.x;
    const int j = threadIdx.x;
    if (j >= ld) return;
    float v = 0.0f;
    if (j < ncols)
        v = fmaxf(g[(size_t)i * ld + j] * scale[j] + shift[j], 0.0f);
    hout[(size_t)i * ld + j] = (bf16_t)v;
    if (write_f32)
        fout[(size_t)i * ld + j] = v;
}

// ---------------------------------------------------------------------------
// Per-event max pool: post-ReLU values are >= 0 so int-bit atomicMax on the
// float bit pattern is order preserving (pooled buffer memset to 0 = +0.0f).
// ---------------------------------------------------------------------------
__global__ void pool_max_kernel(int* __restrict__ pooled_bits,
                                const float* __restrict__ h,
                                const int* __restrict__ batch,
                                int ncols, int ldh, int nrows) {
    int idx = blockIdx.x * blockDim.x + threadIdx.x;
    int i = idx >> 4;
    int j = idx & 15;
    if (i >= nrows || j >= ncols) return;
    float v = h[(size_t)i * ldh + j];
    atomicMax(&pooled_bits[batch[i] * 16 + j], __float_as_int(v));
}

__global__ void head_kernel(float* __restrict__ out, const float* __restrict__ pooled,
                            const float* __restrict__ w, const float* __restrict__ b,
                            int nfeat, int nout) {
    int t = blockIdx.x * blockDim.x + threadIdx.x;
    if (t >= NBATCH * nout) return;
    int bi = t / nout, o = t % nout;
    float s = b[o];
    for (int j = 0; j < nfeat; ++j)
        s += pooled[bi * 16 + j] * w[j * nout + o];
    out[t] = s;
}

// ---------------------------------------------------------------------------
static inline int align32(int x) { return (x + 31) & ~31; }

extern "C" void kernel_launch(void* const* d_in, const int* in_sizes, int n_in,
                              void* d_out, int out_size, void* d_ws, size_t ws_size,
                              hipStream_t stream) {
    static const int PLANES[5] = {256, 195, 134, 73, 10};

    // Input order: top-level dict {x, pos, edge_index, batch, params}; params
    // flattened jax-tree style (dict keys sorted): head{b,w} then layers[l]
    // {b_glb, b_loc, beta, gamma, w_glb, w_loc}.
    const float* x_in  = (const float*)d_in[0];
    const float* pos   = (const float*)d_in[1];
    const int*   eidx  = (const int*)d_in[2];          // [2 x N*K]; row 0 = src
    const int*   batch = (const int*)d_in[3];
    const float* head_b = (const float*)d_in[4];
    const float* head_w = (const float*)d_in[5];
    const int LBASE = 6;
    auto L_bglb  = [&](int l) { return (const float*)d_in[LBASE + 6 * l + 0]; };
    auto L_bloc  = [&](int l) { return (const float*)d_in[LBASE + 6 * l + 1]; };
    auto L_beta  = [&](int l) { return (const float*)d_in[LBASE + 6 * l + 2]; };
    auto L_gamma = [&](int l) { return (const float*)d_in[LBASE + 6 * l + 3]; };
    auto L_wglb  = [&](int l) { return (const float*)d_in[LBASE + 6 * l + 4]; };
    auto L_wloc  = [&](int l) { return (const float*)d_in[LBASE + 6 * l + 5]; };

    // Workspace layout (all offsets 256B aligned).  Npad max = 224.
    char* ws = (char*)d_ws;
    const size_t HBF_OFF  = 0;                                  // bf16 [N x <=256]
    const size_t Y_OFF    = HBF_OFF + (size_t)NNODES * 256 * 2;  // f32 [N x 224]
    const size_t AGG_OFF  = Y_OFF + (size_t)NNODES * 224 * 4;    // f32 [N x 224]
    const size_t WPK1_OFF = AGG_OFF + (size_t)NNODES * 224 * 4;  // packed Wloc
    const size_t WPK2_OFF = WPK1_OFF + 131072;                   // packed Wglb
    const size_t STAT_OFF = WPK2_OFF + 131072;                   // sums/sumsq/scale/shift
    const size_t POOL_OFF = STAT_OFF + 4096;                     // 64 x 16 f32 bits
    bf16_t* Hbf  = (bf16_t*)(ws + HBF_OFF);
    float*  Y    = (float*)(ws + Y_OFF);
    float*  AGG  = (float*)(ws + AGG_OFF);
    bf16_t* Wpk1 = (bf16_t*)(ws + WPK1_OFF);
    bf16_t* Wpk2 = (bf16_t*)(ws + WPK2_OFF);
    float*  sums  = (float*)(ws + STAT_OFF);
    float*  sumsq = sums + 256;
    float*  scale = sums + 512;
    float*  shift = sums + 768;
    int*    pooled = (int*)(ws + POOL_OFF);
    (void)in_sizes; (void)n_in; (void)out_size; (void)ws_size;

    // Layer 0 activation pack: f32 x -> bf16, stride 256 (== F0, K-aligned).
    {
        size_t total = (size_t)NNODES * 256;
        pack_x_kernel<<<(unsigned)((total + 255) / 256), 256, 0, stream>>>(Hbf, x_in, total);
    }

    for (int l = 0; l < 4; ++l) {
        const int nin  = PLANES[l];
        const int nout = PLANES[l + 1];
        const int KpadA = align32(nin);    // A K-dim for local GEMM (== Hbf stride)
        const int Npad  = align32(nout);   // padded output width (also glb-GEMM K)
        const int ktA = KpadA / 32, ktG = Npad / 32, nt16 = Npad / 16;

        // (1) Y = H @ Wloc[:nin]   (no bias / relu yet)
        {
            int total = ktA * nt16 * 512;
            pack_w_kernel<<<(total + 255) / 256, 256, 0, stream>>>(
                Wpk1, L_wloc(l), nin, nout, ktA, nt16);
            dim3 grid((NNODES + 127) / 128, Npad / 32);
            gemm_wmma_bf16<<<grid, 256, 0, stream>>>(
                Hbf, Wpk1, nullptr, Y, NNODES, ktA, nt16, KpadA, Npad, nout, 0);
        }

        // (2) per-edge rank-2 fixup + relu + max aggregation (incl. self loop)
        edge_aggregate_kernel<<<NNODES, 256, 0, stream>>>(
            AGG, Y, pos, eidx /* src row */,
            L_wloc(l) + (size_t)nin * nout /* rows nin, nin+1 */, L_bloc(l),
            nout, Npad);

        // (3) g = relu(agg @ Wglb + b_glb)   (agg -> bf16 first; K == Npad)
        {
            pack_act_kernel<<<NNODES, 256, 0, stream>>>(Hbf, AGG, nout, Npad);
            int total = ktG * nt16 * 512;
            pack_w_kernel<<<(total + 255) / 256, 256, 0, stream>>>(
                Wpk2, L_wglb(l), nout, nout, ktG, nt16);
            dim3 grid((NNODES + 127) / 128, Npad / 32);
            gemm_wmma_bf16<<<grid, 256, 0, stream>>>(
                Hbf, Wpk2, L_bglb(l), Y, NNODES, ktG, nt16, Npad, Npad, nout, 1);
        }

        // (4) BatchNorm (batch statistics) + relu; repack bf16 for next layer
        hipMemsetAsync(sums, 0, 2 * 256 * sizeof(float), stream);
        colstats_kernel<<<(NNODES + 127) / 128, 256, 0, stream>>>(
            Y, sums, sumsq, nout, Npad, 128, NNODES);
        bn_finalize_kernel<<<1, 256, 0, stream>>>(
            sums, sumsq, scale, shift, L_gamma(l), L_beta(l), nout, 1.0f / NNODES);
        // next layer's A stride == align32(nout) == Npad
        bn_relu_pack_kernel<<<NNODES, 256, 0, stream>>>(
            Hbf, AGG, Y, scale, shift, nout, Npad, (l == 3) ? 1 : 0);
    }

    // (5) per-event max pool (final h is in AGG, f32, stride 32) + head
    hipMemsetAsync(pooled, 0, NBATCH * 16 * sizeof(float), stream);
    {
        size_t total = (size_t)NNODES * 16;
        pool_max_kernel<<<(unsigned)((total + 255) / 256), 256, 0, stream>>>(
            pooled, AGG, batch, PLANES[4], 32, NNODES);
    }
    head_kernel<<<1, 256, 0, stream>>>(
        (float*)d_out, (const float*)pooled, head_w, head_b, PLANES[4], 3);
}
